// LowRank2d_22814866276563
// MI455X (gfx1250) — matrix operand order repaired and checked by reading, not compiled, not run on hardware
//
#include <hip/hip_runtime.h>
#include <hip/hip_bf16.h>

// ---------------------------------------------------------------------------
// LowRank2d on MI455X (gfx1250).
// B=8, IN_C=OUT_C=32, S1=S2=256, Nx=65536.
// Fused plan: recompute MLP evals per 16-pt tile with WMMA f16 (never
// materialize phi/psi_eval => mandatory HBM traffic is just 67MB x-read +
// 67MB out-write ~= 5.8us @ 23.3 TB/s; the ~37 GFLOP of MLP work rides on
// v_wmma_f32_16x16x32_f16 with all weights L2-resident).
// Round-4 changes: (a) opaque *offset* (not pointer) so B-fragment loads stay
// un-hoisted but lower as global_load (round 3's opaque pointer killed
// addrspace inference -> flat_load, which also burns DScnt / LDS datapath);
// (b) manually hoist w1/b1/b2/b3 scalars into registers before the tile loop
// (LLVM cannot hoist global loads across __syncthreads); each thread's L1
// column c = t&63 is j-invariant, so L1 needs only 3 resident scalars.
// ---------------------------------------------------------------------------

typedef _Float16 half8  __attribute__((ext_vector_type(8)));
typedef _Float16 half16 __attribute__((ext_vector_type(16)));
typedef float    float8 __attribute__((ext_vector_type(8)));

#define NXTOT 65536
#define NTILES 4096   // NXTOT / 16

union V16 { half16 v; half8 h[2]; };

// A-fragment (16x32 f16, M x K) from row-major LDS matrix, row stride ldk halves.
// lane m=lane&15; lanes<16 take K = kt*32 + {0..15},
// lanes>=16 take K = kt*32 + {16..31} (matches packed B layout).
static __device__ inline half16 load_a_frag(const _Float16* H, int ldk, int lane, int kt) {
    int m    = lane & 15;
    int koff = kt * 32 + ((lane >> 4) << 4);  // +16 for upper half-wave
    const _Float16* p = H + m * ldk + koff;
    V16 u;
    u.h[0] = *(const half8*)(p);
    u.h[1] = *(const half8*)(p + 8);
    return u.v;
}

static __device__ inline float8 fzero8() {
    float8 z = {0.f, 0.f, 0.f, 0.f, 0.f, 0.f, 0.f, 0.f};
    return z;
}

static __device__ inline float8 wmma16x16x32(half16 a, half16 b, float8 c) {
    // (neg_a, A, neg_b, B, c_mod, C, reuse_a, reuse_b)
    return __builtin_amdgcn_wmma_f32_16x16x32_f16(false, a, false, b, (short)0, c, false, false);
}

// Opaque zero: defeats LICM on loads whose address adds this value (so the
// big packed-weight fragment set is re-read from L2 each tile instead of
// being hoisted into VGPRs and spilled), while keeping the base pointer's
// global address space known (=> global_load, not flat_load).
static __device__ inline int opaque_zero() {
    int z = 0;
    asm volatile("" : "+s"(z));
    return z;
}

// Async memory->LDS copy (gfx1250 VGLOBAL async path, tracked by ASYNCcnt).
// VDST VGPR holds the wave-relative LDS byte address (low 32 bits of the
// generic pointer); hardware adds LDS_BASE.
static __device__ inline void async_copy_b128(unsigned lds_addr, const void* gaddr) {
    asm volatile("global_load_async_to_lds_b128 %0, %1, off"
                 :: "v"(lds_addr), "v"(gaddr) : "memory");
}
static __device__ inline void wait_async0() {
    asm volatile("s_wait_asynccnt 0x0" ::: "memory");
}

// ---------------------------------------------------------------------------
// Prep: pack W2 [64x128] and W3 [128x1024] (phi & psi) into f16 fragment-
// native layout: flat index ((kt*NT + nt)*32 + lane)*16 + j  <-  W[k][n],
// k = kt*32 + (lane>=16 ? 16 : 0) + j,  n = nt*16 + (lane&15).
// B-frag load then = one contiguous 32B load per lane (fully coalesced).
// Also zeroes tmp (re-zeroed every launch: deterministic graph replay).
// ---------------------------------------------------------------------------
__global__ __launch_bounds__(256) void lr2d_prep(
    const float* __restrict__ w2a, const float* __restrict__ w3a,
    const float* __restrict__ w2b, const float* __restrict__ w3b,
    _Float16* __restrict__ W2pA, _Float16* __restrict__ W3pA,
    _Float16* __restrict__ W2pB, _Float16* __restrict__ W3pB,
    float* __restrict__ tmp)
{
    int u = blockIdx.x * blockDim.x + threadIdx.x;   // up to 131072
    if (u < 8192) {                                  // W2: kt in 0..1, nt in 0..7
        int j = u & 15, lane = (u >> 4) & 31, nt = (u >> 9) & 7, kt = u >> 12;
        int k = kt * 32 + ((lane >> 4) << 4) + j;
        int n = nt * 16 + (lane & 15);
        W2pA[u] = (_Float16)w2a[k * 128 + n];
        W2pB[u] = (_Float16)w2b[k * 128 + n];
        tmp[u] = 0.0f;                               // tmp is exactly 8192 floats
    }
    if (u < 131072) {                                // W3: kt in 0..3, nt in 0..63
        int j = u & 15, lane = (u >> 4) & 31, nt = (u >> 9) & 63, kt = u >> 15;
        int k = kt * 32 + ((lane >> 4) << 4) + j;
        int n = nt * 16 + (lane & 15);
        W3pA[u] = (_Float16)w3a[k * 1024 + n];
        W3pB[u] = (_Float16)w3b[k * 1024 + n];
    }
}

// ---------------------------------------------------------------------------
// Phase 1: tmp[b,o,i] = sum_n phi_eval[n,o,i] * x[b,i,n]
// Per 16-point tile: L1 (VALU) -> L2 (WMMA, 16x128) -> L3 (WMMA, 16x1024),
// stage P f16 + transposed x-slice in LDS, 32 reg accumulators per thread,
// atomic flush at the end.
// ---------------------------------------------------------------------------
__global__ __launch_bounds__(256, 4) void lr2d_phase1(
    const float* __restrict__ x,
    const float* __restrict__ w1, const float* __restrict__ b1,
    const float* __restrict__ b2, const float* __restrict__ b3,
    const _Float16* __restrict__ W2p, const _Float16* __restrict__ W3p,
    float* __restrict__ tmp)
{
    __shared__ _Float16 H1[16 * 64];     //  2 KB
    __shared__ _Float16 H2[16 * 128];    //  4 KB
    __shared__ _Float16 P16[16 * 1024];  // 32 KB
    __shared__ float    XT[16 * 256];    // 16 KB, transposed: XT[mm][b*32+i]

    const int t = threadIdx.x, lane = t & 31, wave = t >> 5;
    const int mhi = (lane >> 4) << 3;

    // Loop-invariant scalars (must be hoisted manually: __syncthreads is a
    // memory clobber, LLVM won't hoist global loads across it).
    const int c1 = t & 63;                       // L1 column, same for all j
    const float w1x = w1[c1], w1y = w1[64 + c1], b1v = b1[c1];
    const float bias2 = b2[wave * 16 + (lane & 15)];
    float bias3[8];
#pragma unroll
    for (int jj = 0; jj < 8; ++jj)
        bias3[jj] = b3[(wave + jj * 8) * 16 + (lane & 15)];

    float acc[32];
#pragma unroll
    for (int i = 0; i < 32; ++i) acc[i] = 0.0f;

    for (int tile = blockIdx.x; tile < NTILES; tile += gridDim.x) {
        const int base = tile * 16;
        // Per-tile opaque offsets: keep B-fragment loads inside the loop
        // (fresh L2 reads) instead of hoisted+spilled.
        const _Float16* w2t = W2p + opaque_zero();
        const _Float16* w3t = W3p + opaque_zero();

        // ---- L1: H1[16x64] = relu(grid @ W1 + b1), pure VALU ----
#pragma unroll
        for (int j = 0; j < 4; ++j) {
            int m = (t >> 6) + 4 * j;
            int pt = base + m;
            float gx = (float)(pt >> 8)  * (1.0f / 255.0f);
            float gy = (float)(pt & 255) * (1.0f / 255.0f);
            float h = gx * w1x + gy * w1y + b1v;
            H1[m * 64 + c1] = (_Float16)fmaxf(h, 0.0f);
        }
        // ---- stage x slice transposed: XT[mm][t], t = b*32+i ----
        {
            const float* xp = x + (size_t)t * NXTOT + base;
            float vv[16];
#pragma unroll
            for (int q = 0; q < 4; ++q)
                *(float4*)&vv[4 * q] = *(const float4*)(xp + 4 * q);
#pragma unroll
            for (int mm = 0; mm < 16; ++mm)
                XT[mm * 256 + t] = vv[mm];
        }
        __syncthreads();

        // ---- L2: H2[16x128] = relu(H1 @ W2 + b2), wave w owns n-tile w ----
        {
            float8 c8 = fzero8();
#pragma unroll
            for (int kt = 0; kt < 2; ++kt) {
                half16 a = load_a_frag(H1, 64, lane, kt);
                half16 b = *(const half16*)&w2t[((kt * 8 + wave) * 32 + lane) * 16];
                c8 = wmma16x16x32(a, b, c8);
            }
            int n = wave * 16 + (lane & 15);
#pragma unroll
            for (int r = 0; r < 8; ++r)
                H2[(r + mhi) * 128 + n] = (_Float16)fmaxf(c8[r] + bias2, 0.0f);
        }
        __syncthreads();

        // ---- L3: P[16x1024] = H2 @ W3 + b3, wave w owns n-tiles w+8j ----
        {
            half16 afr[4];
#pragma unroll
            for (int kt = 0; kt < 4; ++kt)
                afr[kt] = load_a_frag(H2, 128, lane, kt);
#pragma unroll
            for (int jj = 0; jj < 8; ++jj) {
                int nt = wave + jj * 8;
                float8 c8 = fzero8();
#pragma unroll
                for (int kt = 0; kt < 4; ++kt) {
                    half16 b = *(const half16*)&w3t[((kt * 64 + nt) * 32 + lane) * 16];
                    c8 = wmma16x16x32(afr[kt], b, c8);
                }
                int n = nt * 16 + (lane & 15);
#pragma unroll
                for (int r = 0; r < 8; ++r)
                    P16[(r + mhi) * 1024 + n] = (_Float16)(c8[r] + bias3[jj]);
            }
        }
        __syncthreads();

        // ---- accumulate: thread (b=wave, o=lane):
        //      acc[i] += x[b,i,base+mm] * P[mm][o*32+i]
        // mm-outer keeps the live set small: 8 broadcast b128 x-reads +
        // 4 b128 P-reads + 32 FMA per iteration.
        {
            const int b = wave, o = lane;
            for (int mm = 0; mm < 16; ++mm) {
                float xv[32];
                const float4* xq = (const float4*)&XT[mm * 256 + b * 32];
#pragma unroll
                for (int q = 0; q < 8; ++q)
                    *(float4*)&xv[4 * q] = xq[q];
                half8 ph[4];
                const half8* pp = (const half8*)&P16[mm * 1024 + o * 32];
#pragma unroll
                for (int q = 0; q < 4; ++q)
                    ph[q] = pp[q];
#pragma unroll
                for (int i = 0; i < 32; ++i)
                    acc[i] += xv[i] * (float)ph[i >> 3][i & 7];
            }
        }
        __syncthreads();
    }

#pragma unroll
    for (int i = 0; i < 32; ++i)
        atomicAdd(&tmp[t * 32 + i], acc[i]);   // t = b*32+o here
}

// ---------------------------------------------------------------------------
// Phase 2: out[b,o,m] = (sum_i tmp[b,o,i] * psi_eval[m,o,i]) / Nx
// Same MLP chain; L3 split into two 512-col chunks so LDS fits with the
// block-resident tmp copy (loaded once via the async global->LDS path).
// ---------------------------------------------------------------------------
__global__ __launch_bounds__(256, 4) void lr2d_phase2(
    const float* __restrict__ w1, const float* __restrict__ b1,
    const float* __restrict__ b2, const float* __restrict__ b3,
    const _Float16* __restrict__ W2p, const _Float16* __restrict__ W3p,
    const float* __restrict__ tmp, float* __restrict__ out)
{
    __shared__ _Float16 H1[16 * 64];     //  2 KB
    __shared__ _Float16 H2[16 * 128];    //  4 KB
    __shared__ _Float16 PH[16 * 512];    // 16 KB (half of P columns)
    __shared__ float    TMPL[8192];      // 32 KB  (total 54 KB)

    const int t = threadIdx.x, lane = t & 31, wave = t >> 5;
    const int mhi = (lane >> 4) << 3;

    // Loop-invariant scalars hoisted (see phase1 comment).
    const int c1 = t & 63;
    const float w1x = w1[c1], w1y = w1[64 + c1], b1v = b1[c1];
    const float bias2 = b2[wave * 16 + (lane & 15)];
    float bias3[2][4];
#pragma unroll
    for (int ch = 0; ch < 2; ++ch)
#pragma unroll
        for (int jj = 0; jj < 4; ++jj)
            bias3[ch][jj] = b3[(ch * 32 + wave + jj * 8) * 16 + (lane & 15)];

    // One-time 32KB broadcast of tmp into LDS: async memory->LDS (ASYNCcnt),
    // 128 bytes per thread, no VGPR round-trip.
    {
        unsigned laddr = (unsigned)(size_t)&TMPL[t * 32];
        const char* gp = (const char*)(tmp + t * 32);
#pragma unroll
        for (int q = 0; q < 8; ++q)
            async_copy_b128(laddr + 16 * q, gp + 16 * q);
        wait_async0();
    }
    __syncthreads();

    for (int tile = blockIdx.x; tile < NTILES; tile += gridDim.x) {
        const int base = tile * 16;
        const _Float16* w2t = W2p + opaque_zero();
        const _Float16* w3t = W3p + opaque_zero();

        // ---- L1 ----
#pragma unroll
        for (int j = 0; j < 4; ++j) {
            int m = (t >> 6) + 4 * j;
            int pt = base + m;
            float gx = (float)(pt >> 8)  * (1.0f / 255.0f);
            float gy = (float)(pt & 255) * (1.0f / 255.0f);
            float h = gx * w1x + gy * w1y + b1v;
            H1[m * 64 + c1] = (_Float16)fmaxf(h, 0.0f);
        }
        __syncthreads();

        // ---- L2 ----
        {
            float8 c8 = fzero8();
#pragma unroll
            for (int kt = 0; kt < 2; ++kt) {
                half16 a = load_a_frag(H1, 64, lane, kt);
                half16 b = *(const half16*)&w2t[((kt * 8 + wave) * 32 + lane) * 16];
                c8 = wmma16x16x32(a, b, c8);
            }
            int n = wave * 16 + (lane & 15);
#pragma unroll
            for (int r = 0; r < 8; ++r)
                H2[(r + mhi) * 128 + n] = (_Float16)fmaxf(c8[r] + bias2, 0.0f);
        }
        __syncthreads();

        // ---- L3 + contraction, in two 512-column chunks (o: 16 per chunk) ----
        half16 afr[4];
#pragma unroll
        for (int kt = 0; kt < 4; ++kt)
            afr[kt] = load_a_frag(H2, 128, lane, kt);

        for (int chunk = 0; chunk < 2; ++chunk) {
#pragma unroll
            for (int jj = 0; jj < 4; ++jj) {
                int ntl = wave + jj * 8;             // local n-tile 0..31
                int nt  = chunk * 32 + ntl;
                float8 c8 = fzero8();
#pragma unroll
                for (int kt = 0; kt < 4; ++kt) {
                    half16 b = *(const half16*)&w3t[((kt * 64 + nt) * 32 + lane) * 16];
                    c8 = wmma16x16x32(afr[kt], b, c8);
                }
                int nl = ntl * 16 + (lane & 15);     // local column 0..511
                float bias = (chunk == 0) ? bias3[0][jj] : bias3[1][jj];
#pragma unroll
                for (int r = 0; r < 8; ++r)
                    PH[(r + mhi) * 512 + nl] = (_Float16)(c8[r] + bias);
            }
            __syncthreads();

            // out[b,o,base+m]; thread: m=t&15 (coalesced stores), 8 (b,o) combos
            {
                int m = t & 15, g = t >> 4;
#pragma unroll
                for (int q = 0; q < 8; ++q) {
                    int cc = g * 8 + q;
                    int b = cc >> 4, ol = cc & 15;
                    int o = chunk * 16 + ol;
                    float tv[32];
                    const float4* tq = (const float4*)&TMPL[(b * 32 + o) * 32];
#pragma unroll
                    for (int z = 0; z < 8; ++z)
                        *(float4*)&tv[4 * z] = tq[z];
                    half8 ph[4];
                    const half8* pp = (const half8*)&PH[m * 512 + ol * 32];
#pragma unroll
                    for (int z = 0; z < 4; ++z)
                        ph[z] = pp[z];
                    float s = 0.0f;
#pragma unroll
                    for (int i = 0; i < 32; ++i)
                        s += tv[i] * (float)ph[i >> 3][i & 7];
                    out[(size_t)(b * 32 + o) * NXTOT + base + m] = s * (1.0f / (float)NXTOT);
                }
            }
            __syncthreads();
        }
    }
}

// ---------------------------------------------------------------------------
extern "C" void kernel_launch(void* const* d_in, const int* in_sizes, int n_in,
                              void* d_out, int out_size, void* d_ws, size_t ws_size,
                              hipStream_t stream) {
    const float* x      = (const float*)d_in[0];
    const float* phi_w1 = (const float*)d_in[1];
    const float* phi_b1 = (const float*)d_in[2];
    const float* phi_w2 = (const float*)d_in[3];
    const float* phi_b2 = (const float*)d_in[4];
    const float* phi_w3 = (const float*)d_in[5];
    const float* phi_b3 = (const float*)d_in[6];
    const float* psi_w1 = (const float*)d_in[7];
    const float* psi_b1 = (const float*)d_in[8];
    const float* psi_w2 = (const float*)d_in[9];
    const float* psi_b2 = (const float*)d_in[10];
    const float* psi_w3 = (const float*)d_in[11];
    const float* psi_b3 = (const float*)d_in[12];

    char* ws = (char*)d_ws;
    _Float16* phiW2p = (_Float16*)(ws + 0);       // 16 KB
    _Float16* phiW3p = (_Float16*)(ws + 16384);   // 256 KB
    _Float16* psiW2p = (_Float16*)(ws + 278528);  // 16 KB
    _Float16* psiW3p = (_Float16*)(ws + 294912);  // 256 KB
    float*    tmp    = (float*)   (ws + 557056);  // 32 KB

    lr2d_prep<<<512, 256, 0, stream>>>(phi_w2, phi_w3, psi_w2, psi_w3,
                                       phiW2p, phiW3p, psiW2p, psiW3p, tmp);
    lr2d_phase1<<<256, 256, 0, stream>>>(x, phi_w1, phi_b1, phi_b2, phi_b3,
                                         phiW2p, phiW3p, tmp);
    lr2d_phase2<<<512, 256, 0, stream>>>(psi_w1, psi_b1, psi_b2, psi_b3,
                                         psiW2p, psiW3p, tmp, (float*)d_out);
}
